// Attention_78065325572136
// MI455X (gfx1250) — matrix operand order repaired and checked by reading, not compiled, not run on hardware
//
#include <hip/hip_runtime.h>
#include <hip/hip_bf16.h>
#include <stddef.h>
#include <stdint.h>

// ---------------------------------------------------------------------------
// MI455X (gfx1250) implementation of FFT-attention.
// Strategy: fold rfft/irfft (linear maps) into the projection weights, run
// everything as bf16 WMMA GEMMs + a flash-attention kernel with WMMA.
// Data movement: async global->LDS (ASYNCcnt) when the toolchain exposes it.
// ---------------------------------------------------------------------------

typedef __bf16 v8bf  __attribute__((ext_vector_type(8)));
typedef __bf16 v16bf __attribute__((ext_vector_type(16)));
typedef float  v8f   __attribute__((ext_vector_type(8)));

union F16x16 { v16bf v; v8bf h[2]; };

__device__ __forceinline__ __bf16 f2bf(float f) {
  unsigned u = __builtin_bit_cast(unsigned, f);
  unsigned r = (u + 0x7FFFu + ((u >> 16) & 1u)) >> 16;
  return __builtin_bit_cast(__bf16, (unsigned short)r);
}

#define TWO_PI_OVER_1024 0.0061359231515f  // 2*pi/1024

// ---- async global->LDS (gfx1250 GLOBAL_LOAD_ASYNC_TO_LDS_B128) ------------
#define AS1 __attribute__((address_space(1)))
#define AS3 __attribute__((address_space(3)))

typedef int i32x4 __attribute__((vector_size(16)));  // matches builtin param type

#if defined(__has_builtin)
#if __has_builtin(__builtin_amdgcn_global_load_async_to_lds_b128) && \
    __has_builtin(__builtin_amdgcn_s_wait_asynccnt)
#define USE_ASYNC_LDS 1
#endif
#endif
#ifndef USE_ASYNC_LDS
#define USE_ASYNC_LDS 0
#endif

#if USE_ASYNC_LDS
// per-lane 16-byte async copy; LDS generic ptr low 32 bits == LDS offset (ISA 10.2)
__device__ __forceinline__ void async_cp16(__bf16* lds, const __bf16* g) {
  __builtin_amdgcn_global_load_async_to_lds_b128(
      (AS1 i32x4*)(uintptr_t)g,
      (AS3 i32x4*)(unsigned)(uintptr_t)lds, 0, 0);
}
__device__ __forceinline__ void async_wait0() {
  __builtin_amdgcn_s_wait_asynccnt(0);
}
#endif

// ---------------------------------------------------------------------------
// Pack / generator kernels
// ---------------------------------------------------------------------------

__global__ void pack_f32_bf16(const float* __restrict__ in,
                              __bf16* __restrict__ out, int n) {
  int i = blockIdx.x * blockDim.x + threadIdx.x;
  if (i < n) out[i] = f2bf(in[i]);
}

// wq/wk/wv: [1024 x 1026] f32 -> [1024 x 1056] bf16, zero padded, scaled
__global__ void pack_w_pad(const float* __restrict__ w,
                           __bf16* __restrict__ out, float scale) {
  int i = blockIdx.x * blockDim.x + threadIdx.x;   // 1024*1056
  int row = i / 1056, col = i - row * 1056;
  float v = (col < 1026) ? w[row * 1026 + col] * scale : 0.0f;
  out[i] = f2bf(v);
}

// wproj [f][i] f32 -> bf16 transposed [i][f]
__global__ void pack_wprojT(const float* __restrict__ w,
                            __bf16* __restrict__ out) {
  int i = blockIdx.x * blockDim.x + threadIdx.x;   // 1024*1024
  int r = i >> 10, c = i & 1023;                   // out[r][c] = w[c][r]
  out[i] = f2bf(w[c * 1024 + r]);
}

// Fenc [c][f] : f<513 -> cos(2*pi*f*c/1024)/1024 ; f<1026 -> sin(...)/1024 ; else 0
__global__ void gen_fenc(__bf16* __restrict__ out) {
  int i = blockIdx.x * blockDim.x + threadIdx.x;   // 1024*1056
  int c = i / 1056, f = i - c * 1056;
  float v = 0.0f;
  if (f < 513) {
    int t = (f * c) & 1023;
    v = __cosf((float)t * TWO_PI_OVER_1024) * (1.0f / 1024.0f);
  } else if (f < 1026) {
    int j = f - 513;
    int t = (j * c) & 1023;
    v = __sinf((float)t * TWO_PI_OVER_1024) * (1.0f / 1024.0f);
  }
  out[i] = f2bf(v);
}

// Gt [t][f] : decode matrix transposed.
// f==0 -> 1 ; f<512 -> 2cos(2*pi*f*t/1024) ; f==512 -> 0 ; else 2sin(2*pi*(f-512)t/1024)
__global__ void gen_gt(__bf16* __restrict__ out) {
  int i = blockIdx.x * blockDim.x + threadIdx.x;   // 1024*1024
  int t = i >> 10, f = i & 1023;
  float v;
  if (f < 512) {
    if (f == 0) v = 1.0f;
    else { int a = (f * t) & 1023; v = 2.0f * __cosf((float)a * TWO_PI_OVER_1024); }
  } else {
    int j = f - 512;
    if (j == 0) v = 0.0f;
    else { int a = (j * t) & 1023; v = 2.0f * __sinf((float)a * TWO_PI_OVER_1024); }
  }
  out[i] = f2bf(v);
}

// beff[t] = bproj @ G (f32, kept full precision)
__global__ void gen_beff(const float* __restrict__ bproj,
                         float* __restrict__ beff) {
  int t = blockIdx.x * blockDim.x + threadIdx.x;
  if (t >= 1024) return;
  float acc = bproj[0];
  for (int f = 1; f < 512; ++f) {
    int a = (f * t) & 1023;
    acc += 2.0f * __cosf((float)a * TWO_PI_OVER_1024) * bproj[f];
  }
  for (int j = 1; j < 512; ++j) {
    int a = (j * t) & 1023;
    acc += 2.0f * __sinf((float)a * TWO_PI_OVER_1024) * bproj[512 + j];
  }
  beff[t] = acc;
}

// ---------------------------------------------------------------------------
// bf16 WMMA GEMM:  D[M x N] = A[M x K] * Bt[N x K]^T   (Bt is B transposed,
// row-major [n][k] so staging is layout-preserving and fragment reads are
// contiguous). 128x128 block tile, 8 waves, each wave 32x64, K-step 32.
// Staging: async global->LDS b128 copies when available.
// Epilogue: bf16 store (Dbf) or f32 store + bias (Df).
// ---------------------------------------------------------------------------

__launch_bounds__(256)
__global__ void gemm_bf16_wmma(const __bf16* __restrict__ A, int lda,
                               const __bf16* __restrict__ Bt, int ldb,
                               __bf16* __restrict__ Dbf,
                               float* __restrict__ Df,
                               const float* __restrict__ bias,
                               int M, int N, int K, int ldd) {
  __shared__ __bf16 Alds[128 * 32];
  __shared__ __bf16 Blds[128 * 32];

  const int tid  = threadIdx.x;
  const int lane = tid & 31;
  const int wave = tid >> 5;
  const int m0   = blockIdx.y * 128;
  const int n0   = blockIdx.x * 128;
  const int wm   = (wave & 3) * 32;   // wave M offset
  const int wn   = (wave >> 2) * 64;  // wave N offset

  const int lrow = tid >> 1;          // cooperative-load row 0..127
  const int lcol = (tid & 1) * 16;    // 0 or 16
  const int lm   = lane & 15;
  const int o1   = (lane & 16) ? 8 : 0;   // A-fragment K chunk base
  const int bk   = (lane & 16) ? 16 : 0;  // unused; B fragments also use o1

  v8f acc[2][4];
  for (int i = 0; i < 2; ++i)
    for (int j = 0; j < 4; ++j) acc[i][j] = v8f{};

  for (int k0 = 0; k0 < K; k0 += 32) {
    __syncthreads();
    {
      const __bf16* ga = A + (size_t)(m0 + lrow) * lda + k0 + lcol;
      const __bf16* gb = Bt + (size_t)(n0 + lrow) * ldb + k0 + lcol;
#if USE_ASYNC_LDS
      async_cp16(&Alds[lrow * 32 + lcol],     ga);
      async_cp16(&Alds[lrow * 32 + lcol + 8], ga + 8);
      async_cp16(&Blds[lrow * 32 + lcol],     gb);
      async_cp16(&Blds[lrow * 32 + lcol + 8], gb + 8);
#else
      *(v8bf*)&Alds[lrow * 32 + lcol]     = *(const v8bf*)ga;
      *(v8bf*)&Alds[lrow * 32 + lcol + 8] = *(const v8bf*)(ga + 8);
      *(v8bf*)&Blds[lrow * 32 + lcol]     = *(const v8bf*)gb;
      *(v8bf*)&Blds[lrow * 32 + lcol + 8] = *(const v8bf*)(gb + 8);
#endif
    }
    if (k0 + 32 < K) {  // L2 prefetch of next K tile -> global_prefetch_b8
      __builtin_prefetch(A + (size_t)(m0 + lrow) * lda + k0 + 32 + lcol, 0, 1);
      __builtin_prefetch(Bt + (size_t)(n0 + lrow) * ldb + k0 + 32 + lcol, 0, 1);
    }
#if USE_ASYNC_LDS
    async_wait0();
#endif
    __syncthreads();

    F16x16 af[2];
#pragma unroll
    for (int i = 0; i < 2; ++i) {
      const __bf16* p = &Alds[(wm + i * 16 + lm) * 32];
      af[i].h[0] = *(const v8bf*)(p + o1);
      af[i].h[1] = *(const v8bf*)(p + o1 + 16);
    }
    F16x16 bfm[4];
#pragma unroll
    for (int j = 0; j < 4; ++j) {
      const __bf16* p = &Blds[(wn + j * 16 + lm) * 32];
      bfm[j].h[0] = *(const v8bf*)(p + o1);
      bfm[j].h[1] = *(const v8bf*)(p + o1 + 16);
    }
#pragma unroll
    for (int i = 0; i < 2; ++i)
#pragma unroll
      for (int j = 0; j < 4; ++j)
        acc[i][j] = __builtin_amdgcn_wmma_f32_16x16x32_bf16(
            false, af[i].v, false, bfm[j].v, (short)0, acc[i][j], false, false);
  }

  const int rm = (lane & 16) ? 8 : 0;
#pragma unroll
  for (int i = 0; i < 2; ++i)
#pragma unroll
    for (int j = 0; j < 4; ++j) {
      int mbase = m0 + wm + i * 16 + rm;
      int n = n0 + wn + j * 16 + lm;
#pragma unroll
      for (int r = 0; r < 8; ++r) {
        float v = acc[i][j][r];
        size_t idx = (size_t)(mbase + r) * ldd + n;
        if (Df) Df[idx] = v + (bias ? bias[n] : 0.0f);
        else    Dbf[idx] = f2bf(v);
      }
    }
}

// ---------------------------------------------------------------------------
// Flash attention over the "quirky" head layout:
//   Xhead[m][d] = Xmat[b*1024 + h*64 + m/16][(m%16)*64 + d]  (contig 128B rows)
// Block: 256 thr = 8 waves, 128 Q rows per block (16 per wave), loops over
// 1024 keys in chunks of 32 with online softmax. S = Q K^T and O += P V via
// v_wmma_f32_16x16x32_bf16. K chunk staged async; V chunk transposed manually.
// ---------------------------------------------------------------------------

__launch_bounds__(256)
__global__ void flash_attn(const __bf16* __restrict__ q,
                           const __bf16* __restrict__ k,
                           const __bf16* __restrict__ v,
                           __bf16* __restrict__ omat) {
  __shared__ __bf16 Klds[32 * 64];      // [j][d]  (== Bt layout for QK^T)
  __shared__ __bf16 Vlds[64 * 32];      // [d][j]  (== Bt layout for PV)
  __shared__ __bf16 Pscr[8][16 * 32];   // per-wave P transpose scratch

  const int tid  = threadIdx.x;
  const int lane = tid & 31;
  const int wave = tid >> 5;
  const int bh   = blockIdx.x >> 3;     // b*16 + h
  const int qb   = blockIdx.x & 7;
  const int b    = bh >> 4;
  const int h    = bh & 15;
  const int base_row = b * 1024 + h * 64;
  const int mw   = qb * 128 + wave * 16;
  const int lm   = lane & 15;
  const int o1   = (lane & 16) ? 8 : 0;
  const int bk   = (lane & 16) ? 16 : 0;
  const int rm   = (lane & 16) ? 8 : 0;

  // Q fragments (d = 0..31 and 32..63), loaded once, direct from global
  F16x16 qf[2];
  {
    int mlane = mw + lm;
    const __bf16* qrow =
        q + (size_t)(base_row + (mlane >> 4)) * 1024 + (mlane & 15) * 64;
#pragma unroll
    for (int f = 0; f < 2; ++f) {
      qf[f].h[0] = *(const v8bf*)(qrow + f * 32 + o1);
      qf[f].h[1] = *(const v8bf*)(qrow + f * 32 + o1 + 16);
    }
  }

  float mrow[8], lsum[8];
  v8f oacc[4];
#pragma unroll
  for (int r = 0; r < 8; ++r) { mrow[r] = -1e30f; lsum[r] = 0.0f; }
#pragma unroll
  for (int f = 0; f < 4; ++f) oacc[f] = v8f{};

  for (int j0 = 0; j0 < 1024; j0 += 32) {
    __syncthreads();
    {  // stage K chunk [32][64] (async, layout-preserving) and V transposed
      int j  = tid >> 3;
      int d0 = (tid & 7) * 8;
      int jr = j0 + j;
      size_t grow = (size_t)(base_row + (jr >> 4)) * 1024 + (jr & 15) * 64 + d0;
#if USE_ASYNC_LDS
      async_cp16(&Klds[j * 64 + d0], k + grow);
#else
      *(v8bf*)&Klds[j * 64 + d0] = *(const v8bf*)(k + grow);
#endif
      v8bf vv = *(const v8bf*)(v + grow);
#pragma unroll
      for (int e = 0; e < 8; ++e) Vlds[(d0 + e) * 32 + j] = vv[e];
    }
#if USE_ASYNC_LDS
    async_wait0();
#endif
    __syncthreads();

    // S = Q K^T for 32 key columns (two 16-col C fragments)
    v8f s[2];
#pragma unroll
    for (int nf = 0; nf < 2; ++nf) {
      s[nf] = v8f{};
#pragma unroll
      for (int kf = 0; kf < 2; ++kf) {
        F16x16 kb;
        const __bf16* p = &Klds[(nf * 16 + lm) * 64 + kf * 32];
        kb.h[0] = *(const v8bf*)(p + o1);
        kb.h[1] = *(const v8bf*)(p + o1 + 16);
        s[nf] = __builtin_amdgcn_wmma_f32_16x16x32_bf16(
            false, qf[kf].v, false, kb.v, (short)0, s[nf], false, false);
      }
    }

    // Online softmax; rows live per-VGPR, columns per-lane (width-16 groups)
    __bf16* ps = &Pscr[wave][0];
#pragma unroll
    for (int r = 0; r < 8; ++r) {
      float cmax = fmaxf(s[0][r], s[1][r]);
      for (int msk = 1; msk < 16; msk <<= 1)
        cmax = fmaxf(cmax, __shfl_xor(cmax, msk, 16));
      float mnew = fmaxf(mrow[r], cmax);
      float corr = __expf(mrow[r] - mnew);
      float e0 = __expf(s[0][r] - mnew);
      float e1 = __expf(s[1][r] - mnew);
      float psum = e0 + e1;
      for (int msk = 1; msk < 16; msk <<= 1)
        psum += __shfl_xor(psum, msk, 16);
      lsum[r] = lsum[r] * corr + psum;
      mrow[r] = mnew;
#pragma unroll
      for (int f = 0; f < 4; ++f) oacc[f][r] *= corr;
      int mloc = r + rm;
      ps[mloc * 32 + lm]      = f2bf(e0);
      ps[mloc * 32 + 16 + lm] = f2bf(e1);
    }

    // P: C-layout -> A-layout via per-wave LDS scratch
    F16x16 pf;
    {
      const __bf16* p = &ps[lm * 32];
      pf.h[0] = *(const v8bf*)(p + o1);
      pf.h[1] = *(const v8bf*)(p + o1 + 16);
    }

    // O += P V  (4 d-blocks of 16)
#pragma unroll
    for (int f = 0; f < 4; ++f) {
      F16x16 vb;
      const __bf16* p = &Vlds[(f * 16 + lm) * 32];
      vb.h[0] = *(const v8bf*)(p + o1);
      vb.h[1] = *(const v8bf*)(p + o1 + 16);
      oacc[f] = __builtin_amdgcn_wmma_f32_16x16x32_bf16(
          false, pf.v, false, vb.v, (short)0, oacc[f], false, false);
    }
  }

  // epilogue: normalize, undo head reshape into omat[b*1024+m][h*64+d]
#pragma unroll
  for (int r = 0; r < 8; ++r) {
    float inv = 1.0f / lsum[r];
    int m = mw + rm + r;
#pragma unroll
    for (int f = 0; f < 4; ++f) {
      int d = f * 16 + lm;
      omat[(size_t)(b * 1024 + m) * 1024 + h * 64 + d] = f2bf(oacc[f][r] * inv);
    }
  }
}

// ---------------------------------------------------------------------------
// Launch
// ---------------------------------------------------------------------------

extern "C" void kernel_launch(void* const* d_in, const int* in_sizes, int n_in,
                              void* d_out, int out_size, void* d_ws, size_t ws_size,
                              hipStream_t stream) {
  const float* x     = (const float*)d_in[0];  // [8,1024,1024]
  const float* wq    = (const float*)d_in[1];  // [1024,1026]
  const float* wk    = (const float*)d_in[2];
  const float* wv    = (const float*)d_in[3];
  const float* wproj = (const float*)d_in[4];  // [1024,1024]
  const float* bproj = (const float*)d_in[5];  // [1024]
  float* out = (float*)d_out;                  // [8,1024,1024] f32

  char* ws = (char*)d_ws;
  size_t off = 0;
  auto alloc = [&](size_t bytes) {
    size_t o = off;
    off += (bytes + 255) & ~(size_t)255;
    return o;
  };
  const size_t SZ_X    = (size_t)8192 * 1024 * 2;
  const size_t SZ_WPAD = (size_t)1024 * 1056 * 2;
  const size_t SZ_SQ   = (size_t)1024 * 1024 * 2;

  __bf16* xbf  = (__bf16*)(ws + alloc(SZ_X));
  __bf16* wqb  = (__bf16*)(ws + alloc(SZ_WPAD));
  __bf16* wkb  = (__bf16*)(ws + alloc(SZ_WPAD));
  __bf16* wvb  = (__bf16*)(ws + alloc(SZ_WPAD));
  __bf16* fenc = (__bf16*)(ws + alloc(SZ_WPAD));
  __bf16* gtb  = (__bf16*)(ws + alloc(SZ_SQ));
  __bf16* wpt  = (__bf16*)(ws + alloc(SZ_SQ));
  __bf16* weq  = (__bf16*)(ws + alloc(SZ_SQ));
  __bf16* wek  = (__bf16*)(ws + alloc(SZ_SQ));
  __bf16* wev  = (__bf16*)(ws + alloc(SZ_SQ));
  __bf16* weo  = (__bf16*)(ws + alloc(SZ_SQ));
  __bf16* qm   = (__bf16*)(ws + alloc(SZ_X));
  __bf16* km   = (__bf16*)(ws + alloc(SZ_X));
  __bf16* vm   = (__bf16*)(ws + alloc(SZ_X));
  __bf16* om   = (__bf16*)(ws + alloc(SZ_X));
  float*  beff = (float*)(ws + alloc(1024 * 4));
  (void)ws_size; (void)in_sizes; (void)n_in; (void)out_size;

  // --- pack / generate ---
  pack_f32_bf16<<<(8192 * 1024) / 256, 256, 0, stream>>>(x, xbf, 8192 * 1024);
  pack_w_pad<<<(1024 * 1056) / 256, 256, 0, stream>>>(wq, wqb, 0.125f);  // hd^-0.5
  pack_w_pad<<<(1024 * 1056) / 256, 256, 0, stream>>>(wk, wkb, 1.0f);
  pack_w_pad<<<(1024 * 1056) / 256, 256, 0, stream>>>(wv, wvb, 1.0f);
  pack_wprojT<<<(1024 * 1024) / 256, 256, 0, stream>>>(wproj, wpt);
  gen_fenc<<<(1024 * 1056) / 256, 256, 0, stream>>>(fenc);
  gen_gt<<<(1024 * 1024) / 256, 256, 0, stream>>>(gtb);
  gen_beff<<<4, 256, 0, stream>>>(bproj, beff);

  // --- fold FFT into weights: Weff_*[c_out][c_in] = w* @ Fenc^T (K=1056) ---
  dim3 g1(8, 8), blk(256);
  gemm_bf16_wmma<<<g1, blk, 0, stream>>>(wqb, 1056, fenc, 1056, weq, nullptr,
                                         nullptr, 1024, 1024, 1056, 1024);
  gemm_bf16_wmma<<<g1, blk, 0, stream>>>(wkb, 1056, fenc, 1056, wek, nullptr,
                                         nullptr, 1024, 1024, 1056, 1024);
  gemm_bf16_wmma<<<g1, blk, 0, stream>>>(wvb, 1056, fenc, 1056, wev, nullptr,
                                         nullptr, 1024, 1024, 1056, 1024);
  // Weff_o^T[t][i] = Gt @ wproj (via wproj^T as Bt)
  gemm_bf16_wmma<<<g1, blk, 0, stream>>>(gtb, 1024, wpt, 1024, weo, nullptr,
                                         nullptr, 1024, 1024, 1024, 1024);

  // --- Q/K/V projections: [8192x1024] = x @ Weff ---
  dim3 g2(8, 64);
  gemm_bf16_wmma<<<g2, blk, 0, stream>>>(xbf, 1024, weq, 1024, qm, nullptr,
                                         nullptr, 8192, 1024, 1024, 1024);
  gemm_bf16_wmma<<<g2, blk, 0, stream>>>(xbf, 1024, wek, 1024, km, nullptr,
                                         nullptr, 8192, 1024, 1024, 1024);
  gemm_bf16_wmma<<<g2, blk, 0, stream>>>(xbf, 1024, wev, 1024, vm, nullptr,
                                         nullptr, 8192, 1024, 1024, 1024);

  // --- attention: 128 (b,h) heads x 8 q-blocks ---
  flash_attn<<<1024, 256, 0, stream>>>(qm, km, vm, om);

  // --- output projection + folded irfft + bias, f32 out ---
  gemm_bf16_wmma<<<g2, blk, 0, stream>>>(om, 1024, weo, 1024, nullptr, out,
                                         beff, 8192, 1024, 1024, 1024);
}
